// SelfAttention_87746181857559
// MI455X (gfx1250) — compile-verified, matrix operand order checked by reading, not run on hardware
//
#include <hip/hip_runtime.h>
#include <hip/hip_bf16.h>

typedef __attribute__((ext_vector_type(16))) __bf16 v16bf;
typedef __attribute__((ext_vector_type(8)))  __bf16 v8bf;
typedef __attribute__((ext_vector_type(8)))  float  v8f;

#define LDS_STRIDE 40  // 128x32 bf16 tile, padded row stride (80B) to spread banks

// 32-bit LDS byte offset of a __shared__ object (flat LDS addr = {SHARED_BASE, off}).
__device__ __forceinline__ unsigned lds_off(const void* p) {
  return (unsigned)(unsigned long long)(__attribute__((address_space(3))) const void*)p;
}

// CDNA5 async copy: 16B per lane, global -> LDS, tracked by ASYNCcnt (no VGPR staging).
__device__ __forceinline__ void async_cp16(unsigned lds_dst, const __bf16* gsrc) {
  asm volatile("global_load_async_to_lds_b128 %0, %1, off"
               :
               : "v"(lds_dst), "v"(gsrc)
               : "memory");
}

__device__ __forceinline__ void wait_async0() {
  asm volatile("s_wait_asynccnt 0" ::: "memory");
}

// Load one 16x32 bf16 WMMA operand fragment from an LDS tile stored row-major.
// Per ISA 16-bit A-matrix layout: lanes 0-15 = rows 0-15 with K 0..7 (VGPR0-3) and
// K 16..23 (VGPR4-7); lanes 16-31 = same rows with K 8..15 / 24..31.
__device__ __forceinline__ v16bf frag_ld(const __bf16* base, int lane, int row0) {
  const __bf16* p = base + (row0 + (lane & 15)) * LDS_STRIDE + ((lane >> 4) << 3);
  v8bf lo = *(const v8bf*)(p);        // K = kb .. kb+7
  v8bf hi = *(const v8bf*)(p + 16);   // K = kb+16 .. kb+23
  return __builtin_shufflevector(lo, hi, 0, 1, 2, 3, 4, 5, 6, 7,
                                 8, 9, 10, 11, 12, 13, 14, 15);
}

__device__ __forceinline__ v8f wmma_bf16(v16bf a, v16bf b, v8f c) {
  return __builtin_amdgcn_wmma_f32_16x16x32_bf16(false, a, false, b, (short)0, c,
                                                 false, false);
}

// Generic bf16 GEMM with f32 WMMA accumulation.
//   C[row][col] (or C[col][row] if TRANS_STORE) = scale * sum_k A[row][k]*Bm[col][k] + bias[col]
// A: [M][lda] bf16, Bm: [N][ldb] bf16 (second operand stored n-major => same frag pattern as A).
// Tile 128x128, K-step 32, 256 threads (8 wave32s). Wave grid 4x2; each wave owns a
// 32x64 region = 8 WMMA accumulators. Double-buffered LDS fed by ASYNC global->LDS
// copies: one s_wait_asynccnt + one barrier per K-step; WMMAs overlap the copies.
template <bool TRANS_STORE, bool OUT_BF16>
__global__ void __launch_bounds__(256)
gemm_bf16(const __bf16* __restrict__ A, const __bf16* __restrict__ Bm,
          const float* __restrict__ bias, void* __restrict__ Cout,
          int Kdim, int lda, int ldb, int ldc,
          long sAz, long sBz, long sCz, float scale) {
  __shared__ __align__(16) __bf16 As[2][128 * LDS_STRIDE];
  __shared__ __align__(16) __bf16 Bs[2][128 * LDS_STRIDE];

  const int tid  = threadIdx.x;
  const int lane = tid & 31;
  const int wave = tid >> 5;
  const int wm   = (wave & 3) * 32;   // wave row base within 128 (2 m-tiles)
  const int wn   = (wave >> 2) * 64;  // wave col base within 128 (4 n-tiles)

  const __bf16* Ab = A + (long)blockIdx.z * sAz + (long)(blockIdx.x * 128) * lda;
  const __bf16* Bb = Bm + (long)blockIdx.z * sBz + (long)(blockIdx.y * 128) * ldb;

  // Staging: 128 rows x 32 cols per tile; each of 256 threads moves 16 contiguous bf16
  // (two 16B async copies per matrix).
  const int lr = tid >> 1;         // 0..127
  const int lc = (tid & 1) * 16;   // 0 or 16

  const unsigned la0 = lds_off(&As[0][lr * LDS_STRIDE + lc]);
  const unsigned la1 = lds_off(&As[1][lr * LDS_STRIDE + lc]);
  const unsigned lb0 = lds_off(&Bs[0][lr * LDS_STRIDE + lc]);
  const unsigned lb1 = lds_off(&Bs[1][lr * LDS_STRIDE + lc]);
  const __bf16* ga = Ab + (long)lr * lda + lc;
  const __bf16* gb = Bb + (long)lr * ldb + lc;

  v8f acc[2][4] = {};

  // Prologue: async-stage tile 0 into buffer 0.
  async_cp16(la0, ga);
  async_cp16(la0 + 16, ga + 8);
  async_cp16(lb0, gb);
  async_cp16(lb0 + 16, gb + 8);
  wait_async0();
  __syncthreads();

  const int nk = Kdim >> 5;
  for (int k = 0; k < nk; ++k) {
    const int cur = k & 1;

    // Kick off async copies for tile k+1 into the other buffer; they run in the
    // background while this wave issues its 8 WMMAs.
    if (k + 1 < nk) {
      const int kk = (k + 1) << 5;
      const unsigned lan = cur ? la0 : la1;
      const unsigned lbn = cur ? lb0 : lb1;
      async_cp16(lan, ga + kk);
      async_cp16(lan + 16, ga + kk + 8);
      async_cp16(lbn, gb + kk);
      async_cp16(lbn + 16, gb + kk + 8);
    }
    // Warm GL2 for tile k+2 (global_prefetch_b8).
    if (k + 2 < nk) {
      const int kp = (k + 2) << 5;
      __builtin_prefetch(ga + kp, 0, 1);
      __builtin_prefetch(gb + kp, 0, 1);
    }

    // Load all 6 fragments, then 8 back-to-back WMMAs.
    v16bf af0 = frag_ld(&As[cur][0], lane, wm);
    v16bf af1 = frag_ld(&As[cur][0], lane, wm + 16);
    v16bf bf0 = frag_ld(&Bs[cur][0], lane, wn);
    v16bf bf1 = frag_ld(&Bs[cur][0], lane, wn + 16);
    v16bf bf2 = frag_ld(&Bs[cur][0], lane, wn + 32);
    v16bf bf3 = frag_ld(&Bs[cur][0], lane, wn + 48);
    acc[0][0] = wmma_bf16(af0, bf0, acc[0][0]);
    acc[1][0] = wmma_bf16(af1, bf0, acc[1][0]);
    acc[0][1] = wmma_bf16(af0, bf1, acc[0][1]);
    acc[1][1] = wmma_bf16(af1, bf1, acc[1][1]);
    acc[0][2] = wmma_bf16(af0, bf2, acc[0][2]);
    acc[1][2] = wmma_bf16(af1, bf2, acc[1][2]);
    acc[0][3] = wmma_bf16(af0, bf3, acc[0][3]);
    acc[1][3] = wmma_bf16(af1, bf3, acc[1][3]);

    // Drain the async copies for tile k+1, then make them visible to all waves.
    if (k + 1 < nk) {
      wait_async0();
      __syncthreads();
    }
  }

  // Epilogue. C/D 16x16 f32 layout: VGPR j holds M=j (lanes 0-15) / M=j+8 (lanes 16-31),
  // N = lane & 15.
  const long cz = (long)blockIdx.z * sCz;
#pragma unroll
  for (int mt = 0; mt < 2; ++mt) {
    const int rbase = blockIdx.x * 128 + wm + mt * 16 + ((lane >> 4) << 3);
#pragma unroll
    for (int nt = 0; nt < 4; ++nt) {
      const int col = blockIdx.y * 128 + wn + nt * 16 + (lane & 15);
      const float bcol = bias ? bias[col] : 0.0f;
      v8f acv = acc[mt][nt];
#pragma unroll
      for (int j = 0; j < 8; ++j) {
        const int row = rbase + j;
        float v = acv[j] * scale + bcol;
        long idx = TRANS_STORE ? (cz + (long)col * ldc + row)
                               : (cz + (long)row * ldc + col);
        if (OUT_BF16)
          ((__bf16*)Cout)[idx] = (__bf16)v;
        else
          ((float*)Cout)[idx] = v;
      }
    }
  }
}

// In-place softmax over rows of 2048 bf16. One 256-thread block per row.
__global__ void __launch_bounds__(256)
softmax_rows(__bf16* __restrict__ P, int S) {
  __bf16* p = P + (long)blockIdx.x * S;
  const int tid  = threadIdx.x;
  const int lane = tid & 31;
  const int wave = tid >> 5;
  __shared__ float red[8];

  float vals[8];
  float m = -3.0e38f;
#pragma unroll
  for (int i = 0; i < 8; ++i) {
    float v = (float)p[tid + i * 256];
    vals[i] = v;
    m = fmaxf(m, v);
  }
#pragma unroll
  for (int off = 16; off > 0; off >>= 1) m = fmaxf(m, __shfl_xor(m, off, 32));
  if (lane == 0) red[wave] = m;
  __syncthreads();
  m = red[0];
#pragma unroll
  for (int w = 1; w < 8; ++w) m = fmaxf(m, red[w]);
  __syncthreads();

  float s = 0.0f;
#pragma unroll
  for (int i = 0; i < 8; ++i) {
    vals[i] = __expf(vals[i] - m);
    s += vals[i];
  }
#pragma unroll
  for (int off = 16; off > 0; off >>= 1) s += __shfl_xor(s, off, 32);
  if (lane == 0) red[wave] = s;
  __syncthreads();
  s = red[0];
#pragma unroll
  for (int w = 1; w < 8; ++w) s += red[w];
  const float inv = 1.0f / s;
#pragma unroll
  for (int i = 0; i < 8; ++i) p[tid + i * 256] = (__bf16)(vals[i] * inv);
}

__global__ void cvt_f32_bf16(const float* __restrict__ in,
                             __bf16* __restrict__ out, long n) {
  long i = (long)blockIdx.x * blockDim.x + threadIdx.x;
  const long stride = (long)gridDim.x * blockDim.x;
  for (; i < n; i += stride) out[i] = (__bf16)in[i];
}

extern "C" void kernel_launch(void* const* d_in, const int* in_sizes, int n_in,
                              void* d_out, int out_size, void* d_ws, size_t ws_size,
                              hipStream_t stream) {
  (void)in_sizes; (void)n_in; (void)out_size; (void)ws_size;
  const int B = 4, S = 2048, D = 1024;

  const float* x  = (const float*)d_in[0];
  const float* wq = (const float*)d_in[1];
  const float* bq = (const float*)d_in[2];
  const float* wk = (const float*)d_in[3];
  const float* bk = (const float*)d_in[4];
  const float* wv = (const float*)d_in[5];
  const float* bv = (const float*)d_in[6];
  float* out = (float*)d_out;

  const long nX = (long)B * S * D;  // 8,388,608
  const long nW = (long)D * D;      // 1,048,576
  const long nP = (long)B * S * S;  // 16,777,216

  // Workspace carve (all bf16): ~107 MB total, L2-resident on MI455X (192 MB L2).
  char* w = (char*)d_ws;
  __bf16* xb  = (__bf16*)w;  w += nX * 2;
  __bf16* wqb = (__bf16*)w;  w += nW * 2;
  __bf16* wkb = (__bf16*)w;  w += nW * 2;
  __bf16* wvb = (__bf16*)w;  w += nW * 2;
  __bf16* qb  = (__bf16*)w;  w += nX * 2;  // [b][s][e]
  __bf16* kb  = (__bf16*)w;  w += nX * 2;  // [b][s][e]
  __bf16* vtb = (__bf16*)w;  w += nX * 2;  // [b][e][s]  (transposed for P@V)
  __bf16* pb  = (__bf16*)w;  w += nP * 2;  // [b][sq][sk]

  cvt_f32_bf16<<<2048, 256, 0, stream>>>(x, xb, nX);
  cvt_f32_bf16<<<512, 256, 0, stream>>>(wq, wqb, nW);
  cvt_f32_bf16<<<512, 256, 0, stream>>>(wk, wkb, nW);
  cvt_f32_bf16<<<512, 256, 0, stream>>>(wv, wvb, nW);

  const dim3 blk(256);

  // QKV projections: out[b,s,e] = sum_d x[b,s,d]*W[e,d] + bias[e]
  const dim3 gQKV(S / 128, D / 128, B);
  gemm_bf16<false, true><<<gQKV, blk, 0, stream>>>(
      xb, wqb, bq, qb, D, D, D, D, (long)S * D, 0L, (long)S * D, 1.0f);
  gemm_bf16<false, true><<<gQKV, blk, 0, stream>>>(
      xb, wkb, bk, kb, D, D, D, D, (long)S * D, 0L, (long)S * D, 1.0f);
  gemm_bf16<true, true><<<gQKV, blk, 0, stream>>>(  // V stored [d][s]
      xb, wvb, bv, vtb, D, D, D, S, (long)S * D, 0L, (long)D * S, 1.0f);

  // scores[b,sq,sk] = (Q . K) / sqrt(D);  1/sqrt(1024) = 0.03125
  const dim3 gS(S / 128, S / 128, B);
  gemm_bf16<false, true><<<gS, blk, 0, stream>>>(
      qb, kb, nullptr, pb, D, D, D, S, (long)S * D, (long)S * D, (long)S * S,
      0.03125f);

  softmax_rows<<<B * S, 256, 0, stream>>>(pb, S);

  // out[b,sq,d] = sum_sk P[b,sq,sk] * V[b,sk,d]  (B-operand = Vt rows n=d)
  const dim3 gO(S / 128, D / 128, B);
  gemm_bf16<false, false><<<gO, blk, 0, stream>>>(
      pb, vtb, nullptr, out, S, S, S, D, (long)S * S, (long)D * S, (long)S * D,
      1.0f);
}
